// NeighborCell_88562225644176
// MI455X (gfx1250) — matrix operand order repaired and checked by reading, not compiled, not run on hardware
//
#include <hip/hip_runtime.h>
#include <hip/hip_bf16.h>
#include <math.h>

// ---------------------------------------------------------------------------
// Problem constants (match the reference)
// ---------------------------------------------------------------------------
#define NROWS  65536   // N: total neighbor rows
#define BSAMP  4096    // B: samples (segments)
#define IN_DIM 64
#define H_DIM  512
#define D_DIM  256
#define EMB_IN 832     // IN + H + D
#define G3     1536    // 3*H

#define LDP    40      // LDS row pitch (32 data + 8 pad) to avoid bank conflicts
#define KSTEP  32      // bf16 WMMA K

typedef __attribute__((ext_vector_type(16))) __bf16 v16bf;
typedef __attribute__((ext_vector_type(8)))  __bf16 v8bf;
typedef __attribute__((ext_vector_type(8)))  float  v8f;

// Native f32 -> bf16 (compiler emits v_cvt_pk_bf16_f32 pairs on gfx1250)
__device__ __forceinline__ __bf16 f2bf(float f) { return (__bf16)f; }

// Assemble a 16-element bf16 fragment from two contiguous 8-element chunks.
__device__ __forceinline__ v16bf ld2x8(const __bf16* p0, const __bf16* p1) {
    v8bf lo = *(const v8bf*)p0;
    v8bf hi = *(const v8bf*)p1;
    return __builtin_shufflevector(lo, hi, 0,1,2,3,4,5,6,7,8,9,10,11,12,13,14,15);
}

__device__ __forceinline__ v8f wmma_bf16(v16bf a, v16bf b, v8f c) {
    return __builtin_amdgcn_wmma_f32_16x16x32_bf16(
        /*neg_a=*/false, a, /*neg_b=*/false, b,
        /*c_mod=*/(short)0, c, /*reuse_a=*/false, /*reuse_b=*/false);
}

// Async global -> LDS copy of 16 bytes (CDNA5, tracked by ASYNCcnt).
// GVS form: 64-bit SGPR base + signed 32-bit VGPR byte offset.
// LDS address VGPR = low 32 bits of the generic pointer to a __shared__ object.
__device__ __forceinline__ void async_b128(const __bf16* sbase, unsigned byteOff,
                                           const __bf16* ldsp) {
    unsigned lds_addr = (unsigned)(uintptr_t)ldsp;
    asm volatile("global_load_async_to_lds_b128 %0, %1, %2"
                 :: "v"(lds_addr), "v"(byteOff), "s"(sbase) : "memory");
}
#define S_WAIT_ASYNCCNT(N) asm volatile("s_wait_asynccnt " #N ::: "memory")

// ---------------------------------------------------------------------------
// Prep kernels
// ---------------------------------------------------------------------------
// sample_id[n] = searchsorted(starts, n, 'right') - 1   (starts sorted, starts[0]==0)
__global__ void k_sample_id(const int* __restrict__ starts, int* __restrict__ sid) {
    int n = blockIdx.x * 256 + threadIdx.x;
    if (n >= NROWS) return;
    int lo = 0, hi = BSAMP;
    while (lo < hi) {
        int mid = (lo + hi) >> 1;
        if (starts[mid] <= n) lo = mid + 1; else hi = mid;
    }
    sid[n] = lo - 1;
}

// Vectorized f32 -> bf16 conversion, 8 elements per thread (n must be /8)
__global__ void k_cvt_bf16_v8(const float* __restrict__ s, __bf16* __restrict__ d, int n8) {
    int i = blockIdx.x * 256 + threadIdx.x;
    if (i >= n8) return;
    const float4* sp = (const float4*)s + (size_t)i * 2;
    float4 a = sp[0], b = sp[1];
    v8bf o;
    o[0] = f2bf(a.x); o[1] = f2bf(a.y); o[2] = f2bf(a.z); o[3] = f2bf(a.w);
    o[4] = f2bf(b.x); o[5] = f2bf(b.y); o[6] = f2bf(b.z); o[7] = f2bf(b.w);
    *(v8bf*)(d + (size_t)i * 8) = o;
}

// W_embT[j][k] = W_emb[k][j]  (columns of W_emb become contiguous K rows)
__global__ void k_transpose_wemb(const float* __restrict__ w, __bf16* __restrict__ wt) {
    int i = blockIdx.x * 256 + threadIdx.x;
    if (i >= H_DIM * EMB_IN) return;
    int j = i / EMB_IN;
    int k = i - j * EMB_IN;
    wt[i] = f2bf(w[(size_t)k * H_DIM + j]);
}

// ---------------------------------------------------------------------------
// Pass 1: emb = relu(concat(neighbor_t, traj[sid], dist) @ W_emb + b_emb)
// Block: 256 threads (8 waves), tile 128 rows x 64 cols. Wave: 32x32.
// X tile staged via VALU (needs f32->bf16 + ragged concat); W tile staged via
// async global->LDS loads that overlap the X conversion work.
// ---------------------------------------------------------------------------
__launch_bounds__(256)
__global__ void k_gemm_emb(const float* __restrict__ neighbor_t,
                           const float* __restrict__ traj,
                           const float* __restrict__ dist,
                           const __bf16* __restrict__ wembT,   // [512][832]
                           const float* __restrict__ b_emb,
                           const int* __restrict__ sid,
                           __bf16* __restrict__ emb)           // [N][512] bf16
{
    __shared__ __attribute__((aligned(32))) __bf16 Xs[128 * LDP];
    __shared__ __attribute__((aligned(32))) __bf16 Ws[64 * LDP];

    const int tid  = threadIdx.x;
    const int lane = tid & 31;
    const int wave = tid >> 5;
    const int wy   = wave & 3;   // 4 row groups of 32
    const int wx   = wave >> 2;  // 2 col groups of 32
    const int rowBase = blockIdx.x * 128;
    const int colBase = blockIdx.y * 64;

    v8f acc[2][2] = {};

    for (int k0 = 0; k0 < EMB_IN; k0 += KSTEP) {
        // ---- async-stage W tile [64 cols x 32 k]: 256 b128 units, 1/thread ----
        {
            int r = tid >> 2, q = tid & 3;
            async_b128(wembT, (unsigned)((((colBase + r) * EMB_IN) + k0 + q * 8) * 2),
                       &Ws[r * LDP + q * 8]);
        }
        // ---- stage X tile [128 x 32] with ragged-concat + f32->bf16 ----
        {
            int row  = tid >> 1;          // 2 threads per row
            int half = tid & 1;
            int n    = rowBase + row;
            int kk   = k0 + half * 16;
            const float* src;
            if (k0 < IN_DIM) {
                src = neighbor_t + (size_t)n * IN_DIM + kk;
            } else if (k0 < IN_DIM + H_DIM) {
                src = traj + (size_t)sid[n] * H_DIM + (kk - IN_DIM);
            } else {
                src = dist + (size_t)n * D_DIM + (kk - IN_DIM - H_DIM);
            }
            __bf16* dp = &Xs[row * LDP + half * 16];
            #pragma unroll
            for (int i = 0; i < 16; i++) dp[i] = f2bf(src[i]);
        }
        S_WAIT_ASYNCCNT(0);
        __syncthreads();

        const int m    = lane & 15;
        const int half = lane >> 4;
        v16bf a[2], b[2];
        #pragma unroll
        for (int i = 0; i < 2; i++) {
            const __bf16* p = &Xs[(wy * 32 + i * 16 + m) * LDP + half * 16];
            a[i] = ld2x8(p, p + 8);
        }
        #pragma unroll
        for (int j = 0; j < 2; j++) {
            const __bf16* p = &Ws[(wx * 32 + j * 16 + m) * LDP + half * 16];
            b[j] = ld2x8(p, p + 8);
        }
        #pragma unroll
        for (int i = 0; i < 2; i++)
            #pragma unroll
            for (int j = 0; j < 2; j++)
                acc[i][j] = wmma_bf16(a[i], b[j], acc[i][j]);
        __syncthreads();
    }

    // ---- epilogue: +bias, relu, store bf16 ----
    const int m    = lane & 15;
    const int half = lane >> 4;
    #pragma unroll
    for (int i = 0; i < 2; i++) {
        #pragma unroll
        for (int j = 0; j < 2; j++) {
            int col = colBase + wx * 32 + j * 16 + m;
            float bias = b_emb[col];
            #pragma unroll
            for (int r = 0; r < 8; r++) {
                int row = rowBase + wy * 32 + i * 16 + r + half * 8;
                float v = acc[i][j][r] + bias;
                v = v > 0.f ? v : 0.f;
                emb[(size_t)row * H_DIM + col] = f2bf(v);
            }
        }
    }
}

// ---------------------------------------------------------------------------
// Pass 2: fused GRUCell with async double-buffered LDS staging.
// Block tile = 128 rows x 32 out cols; six gate GEMMs (i_r,i_z,i_n / h_r,h_z,h_n)
// accumulate over K=512, then gate math + blend with f32 ht.
// Per k-step each thread issues exactly 7 async b128 loads (1792 units total):
//   [0,512) Es  [512,1024) Hs  [1024,1408) Wi  [1408,1792) Wh
// (all branch boundaries are multiples of 32 -> waves stay uniform).
// ---------------------------------------------------------------------------
__launch_bounds__(256)
__global__ void k_gru(const __bf16* __restrict__ emb,   // [N][512]
                      const __bf16* __restrict__ htb,   // [N][512] bf16
                      const __bf16* __restrict__ wih,   // [1536][512] bf16
                      const __bf16* __restrict__ whh,   // [1536][512] bf16
                      const float*  __restrict__ b_ih,
                      const float*  __restrict__ b_hh,
                      const float*  __restrict__ ht32,  // original f32 ht
                      float* __restrict__ out)
{
    __shared__ __attribute__((aligned(32))) __bf16 Es[2][128 * LDP];
    __shared__ __attribute__((aligned(32))) __bf16 Hs[2][128 * LDP];
    __shared__ __attribute__((aligned(32))) __bf16 Wi[2][96 * LDP];
    __shared__ __attribute__((aligned(32))) __bf16 Wh[2][96 * LDP];

    const int tid  = threadIdx.x;
    const int lane = tid & 31;
    const int wave = tid >> 5;
    const int wy   = wave & 3;   // 4 row groups of 32
    const int wx   = wave >> 2;  // 2 col groups of 16
    const int rowBase = blockIdx.x * 128;
    const int colBase = blockIdx.y * 32;

    // issue the 7 async b128 loads for k-step k0 into buffer `buf`
    auto prefetch = [&](int buf, int k0) {
        #pragma unroll
        for (int i = 0; i < 7; i++) {
            int u = tid + i * 256;
            if (u < 512) {
                int r = u >> 2, q = u & 3;
                async_b128(emb, (unsigned)((((rowBase + r) * H_DIM) + k0 + q * 8) * 2),
                           &Es[buf][r * LDP + q * 8]);
            } else if (u < 1024) {
                int v = u - 512;
                int r = v >> 2, q = v & 3;
                async_b128(htb, (unsigned)((((rowBase + r) * H_DIM) + k0 + q * 8) * 2),
                           &Hs[buf][r * LDP + q * 8]);
            } else if (u < 1408) {
                int v = u - 1024;
                int r = v >> 2, q = v & 3;
                int g = r >> 5, cl = r & 31;
                async_b128(wih, (unsigned)((((g * H_DIM + colBase + cl) * H_DIM) + k0 + q * 8) * 2),
                           &Wi[buf][r * LDP + q * 8]);
            } else {
                int v = u - 1408;
                int r = v >> 2, q = v & 3;
                int g = r >> 5, cl = r & 31;
                async_b128(whh, (unsigned)((((g * H_DIM + colBase + cl) * H_DIM) + k0 + q * 8) * 2),
                           &Wh[buf][r * LDP + q * 8]);
            }
        }
    };

    v8f acc[2][6] = {};   // [row tile][i_r,i_z,i_n,h_r,h_z,h_n]

    prefetch(0, 0);
    int cur = 0;
    for (int k0 = 0; k0 < H_DIM; k0 += KSTEP, cur ^= 1) {
        if (k0 + KSTEP < H_DIM) {
            prefetch(cur ^ 1, k0 + KSTEP);  // overlap next tile with this compute
            // async loads retire in order: cnt<=7 means the 7 current-tile loads done
            S_WAIT_ASYNCCNT(7);
        } else {
            S_WAIT_ASYNCCNT(0);
        }
        __syncthreads();

        const int m    = lane & 15;
        const int half = lane >> 4;
        v16bf ae[2], ah[2], bi[3], bh[3];
        #pragma unroll
        for (int i = 0; i < 2; i++) {
            const __bf16* pe = &Es[cur][(wy * 32 + i * 16 + m) * LDP + half * 16];
            const __bf16* ph = &Hs[cur][(wy * 32 + i * 16 + m) * LDP + half * 16];
            ae[i] = ld2x8(pe, pe + 8);
            ah[i] = ld2x8(ph, ph + 8);
        }
        #pragma unroll
        for (int g = 0; g < 3; g++) {
            const __bf16* pi = &Wi[cur][(g * 32 + wx * 16 + m) * LDP + half * 16];
            const __bf16* ph = &Wh[cur][(g * 32 + wx * 16 + m) * LDP + half * 16];
            bi[g] = ld2x8(pi, pi + 8);
            bh[g] = ld2x8(ph, ph + 8);
        }
        #pragma unroll
        for (int i = 0; i < 2; i++) {
            #pragma unroll
            for (int g = 0; g < 3; g++) {
                acc[i][g]     = wmma_bf16(ae[i], bi[g], acc[i][g]);
                acc[i][3 + g] = wmma_bf16(ah[i], bh[g], acc[i][3 + g]);
            }
        }
        __syncthreads();
    }

    // ---- gate math + blend ----
    const int m    = lane & 15;
    const int half = lane >> 4;
    const int col  = colBase + wx * 16 + m;
    const float bir = b_ih[col], biz = b_ih[H_DIM + col], bin = b_ih[2 * H_DIM + col];
    const float bhr = b_hh[col], bhz = b_hh[H_DIM + col], bhn = b_hh[2 * H_DIM + col];
    #pragma unroll
    for (int i = 0; i < 2; i++) {
        #pragma unroll
        for (int r = 0; r < 8; r++) {
            int row = rowBase + wy * 32 + i * 16 + r + half * 8;
            float ir  = acc[i][0][r] + bir;
            float iz  = acc[i][1][r] + biz;
            float in_ = acc[i][2][r] + bin;
            float hr  = acc[i][3][r] + bhr;
            float hz  = acc[i][4][r] + bhz;
            float hn  = acc[i][5][r] + bhn;
            float rg = 1.f / (1.f + __expf(-(ir + hr)));
            float z  = 1.f / (1.f + __expf(-(iz + hz)));
            float nn = tanhf(in_ + rg * hn);
            float h  = ht32[(size_t)row * H_DIM + col];
            out[(size_t)row * H_DIM + col] = (1.f - z) * nn + z * h;
        }
    }
}

// ---------------------------------------------------------------------------
// Host launcher
// ---------------------------------------------------------------------------
extern "C" void kernel_launch(void* const* d_in, const int* in_sizes, int n_in,
                              void* d_out, int out_size, void* d_ws, size_t ws_size,
                              hipStream_t stream)
{
    const float* traj    = (const float*)d_in[0];   // [B,H]
    const float* neigh   = (const float*)d_in[1];   // [N,IN]
    const float* dist    = (const float*)d_in[2];   // [N,D]
    const float* ht      = (const float*)d_in[3];   // [N,H]
    const float* W_emb   = (const float*)d_in[4];   // [832,512]
    const float* b_emb   = (const float*)d_in[5];
    const float* w_ih    = (const float*)d_in[6];   // [1536,512]
    const float* w_hh    = (const float*)d_in[7];   // [1536,512]
    const float* b_ih    = (const float*)d_in[8];
    const float* b_hh    = (const float*)d_in[9];
    const int*   starts  = (const int*)d_in[10];    // [B]
    float*       out     = (float*)d_out;

    // Workspace layout (256B-aligned), total ~132 MB
    char* ws = (char*)d_ws;
    int*    sid   = (int*)   (ws + 0);                       // 256 KB
    __bf16* wembT = (__bf16*)(ws + 262144);                  // 512*832*2
    __bf16* wihb  = (__bf16*)(ws + 1114112);                 // 1536*512*2
    __bf16* whhb  = (__bf16*)(ws + 2686976);                 // 1536*512*2
    __bf16* htb   = (__bf16*)(ws + 4259840);                 // 65536*512*2
    __bf16* embb  = (__bf16*)(ws + 71368704);                // 65536*512*2
    (void)ws_size; (void)in_sizes; (void)n_in; (void)out_size;

    // Prep
    k_sample_id<<<NROWS / 256, 256, 0, stream>>>(starts, sid);
    k_transpose_wemb<<<(H_DIM * EMB_IN + 255) / 256, 256, 0, stream>>>(W_emb, wembT);
    k_cvt_bf16_v8<<<(G3 * H_DIM / 8 + 255) / 256, 256, 0, stream>>>(w_ih, wihb, G3 * H_DIM / 8);
    k_cvt_bf16_v8<<<(G3 * H_DIM / 8 + 255) / 256, 256, 0, stream>>>(w_hh, whhb, G3 * H_DIM / 8);
    k_cvt_bf16_v8<<<(NROWS * H_DIM / 8) / 256, 256, 0, stream>>>(ht, htb, NROWS * H_DIM / 8);

    // Pass 1: emb GEMM (N x 512 over K=832)
    dim3 g1(NROWS / 128, H_DIM / 64);
    k_gemm_emb<<<g1, 256, 0, stream>>>(neigh, traj, dist, wembT, b_emb, sid, embb);

    // Pass 2: fused GRU (N x 512, six gate GEMMs over K=512 + elementwise)
    dim3 g2(NROWS / 128, H_DIM / 32);
    k_gru<<<g2, 256, 0, stream>>>(embb, htb, wihb, whhb, b_ih, b_hh, ht, out);
}
